// SpectralEncoder_82411832476341
// MI455X (gfx1250) — matrix operand order repaired and checked by reading, not compiled, not run on hardware
//
#include <hip/hip_runtime.h>

typedef _Float16 f16;
typedef __attribute__((ext_vector_type(16))) _Float16 v16h;
typedef __attribute__((ext_vector_type(8)))  float    v8f;

// ---------------- sparse / elementwise kernels ----------------

__global__ void deg_kernel(const long long* __restrict__ src,
                           const float* __restrict__ ew,
                           float* __restrict__ deg, int E) {
    int e = blockIdx.x * blockDim.x + threadIdx.x;
    if (e < E) atomicAdd(&deg[(int)src[e]], ew[e]);
}

__global__ void edgew_kernel(const long long* __restrict__ src,
                             const long long* __restrict__ dst,
                             const float* __restrict__ ew,
                             const float* __restrict__ deg,
                             float* __restrict__ w, int E) {
    int e = blockIdx.x * blockDim.x + threadIdx.x;
    if (e >= E) return;
    float ds = deg[(int)src[e]];
    float dd = deg[(int)dst[e]];
    float is = ds > 0.f ? rsqrtf(fmaxf(ds, 1e-12f)) : 0.f;
    float id = dd > 0.f ? rsqrtf(fmaxf(dd, 1e-12f)) : 0.f;
    w[e] = -is * ew[e] * id;
}

// h[n,0:112]=x, h[n,112:128]=lap_pe ; also seed Acat slot0 (f16)
__global__ void concat_kernel(const float* __restrict__ x,
                              const float* __restrict__ pe,
                              float* __restrict__ h,
                              f16* __restrict__ Acat, int N) {
    int t = blockIdx.x * blockDim.x + threadIdx.x;
    int n = t >> 5, g = t & 31;
    if (n >= N) return;
    float4 v;
    if (g < 28) v = *(const float4*)(x + (size_t)n * 112 + g * 4);
    else        v = *(const float4*)(pe + (size_t)n * 16 + (g - 28) * 4);
    *(float4*)(h + (size_t)n * 128 + g * 4) = v;
    f16* a = Acat + (size_t)n * 512 + g * 4;
    a[0] = (f16)v.x; a[1] = (f16)v.y; a[2] = (f16)v.z; a[3] = (f16)v.w;
}

// fp32 [N,128] -> f16 into Acat column-slot `slot` (K-concat layout, lda=512)
__global__ void cvt_kernel(const float* __restrict__ hsrc,
                           f16* __restrict__ Acat, int slot, int N) {
    int t = blockIdx.x * blockDim.x + threadIdx.x;
    int n = t >> 5, g = t & 31;
    if (n >= N) return;
    float4 v = *(const float4*)(hsrc + (size_t)n * 128 + g * 4);
    f16* a = Acat + (size_t)n * 512 + slot * 128 + g * 4;
    a[0] = (f16)v.x; a[1] = (f16)v.y; a[2] = (f16)v.z; a[3] = (f16)v.w;
}

// P = scatter_add(w[e] * T[src[e]] -> dst[e]) ; one wave per edge, 4 floats/lane
__global__ void prop_kernel(const float* __restrict__ T,
                            const float* __restrict__ w,
                            const long long* __restrict__ src,
                            const long long* __restrict__ dst,
                            float* __restrict__ P, int E) {
    int t = blockIdx.x * blockDim.x + threadIdx.x;
    int e = t >> 5, lane = t & 31;
    if (e >= E) return;
    int s = (int)src[e], d = (int)dst[e];
    float we = w[e];
    float4 v = *(const float4*)(T + (size_t)s * 128 + lane * 4);
    float* pd = P + (size_t)d * 128 + lane * 4;
    atomicAdd(pd + 0, we * v.x);
    atomicAdd(pd + 1, we * v.y);
    atomicAdd(pd + 2, we * v.z);
    atomicAdd(pd + 3, we * v.w);
}

// Tnew = alpha*P + beta*Told ; optional fp32 store ; always writes f16 slot of Acat
__global__ void cheb_kernel(const float* __restrict__ P,
                            const float* __restrict__ Told,
                            float alpha, float beta,
                            float* __restrict__ Tnew,
                            f16* __restrict__ Acat, int slot, int N) {
    int t = blockIdx.x * blockDim.x + threadIdx.x;
    int n = t >> 5, g = t & 31;
    if (n >= N) return;
    float4 p = *(const float4*)(P + (size_t)n * 128 + g * 4);
    float4 o = *(const float4*)(Told + (size_t)n * 128 + g * 4);
    float4 v;
    v.x = alpha * p.x + beta * o.x;
    v.y = alpha * p.y + beta * o.y;
    v.z = alpha * p.z + beta * o.z;
    v.w = alpha * p.w + beta * o.w;
    if (Tnew) *(float4*)(Tnew + (size_t)n * 128 + g * 4) = v;
    f16* a = Acat + (size_t)n * 512 + slot * 128 + g * 4;
    a[0] = (f16)v.x; a[1] = (f16)v.y; a[2] = (f16)v.z; a[3] = (f16)v.w;
}

// Pre-swizzle B [Kdim,128] (fp32 row-major) into WMMA 32x16-tile lane layout:
// Bsw[kt][nt][lane][j] = B[kt*32 + (lane&16) + j, nt*16 + (lane&15)]
// If nsplit>0: cols<nsplit come from W0 [Kdim,nsplit], rest from W1 [Kdim,nsplit].
__global__ void packb_kernel(const float* __restrict__ W0,
                             const float* __restrict__ W1,
                             int nsplit, f16* __restrict__ Bsw, int Kdim) {
    int t = blockIdx.x * blockDim.x + threadIdx.x;
    if (t >= Kdim * 128) return;
    int j    = t & 15;
    int lane = (t >> 4) & 31;
    int nt   = (t >> 9) & 7;
    int kt   = t >> 12;
    int K  = kt * 32 + (lane & 16) + j;
    int nc = nt * 16 + (lane & 15);
    float val;
    if (nsplit > 0)
        val = (nc < nsplit) ? W0[(size_t)K * nsplit + nc]
                            : W1[(size_t)K * nsplit + (nc - nsplit)];
    else
        val = W0[(size_t)K * 128 + nc];
    Bsw[t] = (f16)val;
}

// ---------------- WMMA GEMM: C[N,128] = A[N,512(f16)] @ Bsw + bias ----------------
// 256 threads = 8 waves; block tile 64 rows x 128 cols; wave = 16 rows x 64 cols
// flags: bit0 = relu, bit1 = split-store (cols 0:64 -> out[.,64], 64:128 -> out+N*64)
__global__ void __launch_bounds__(256)
gemm_wmma(const f16* __restrict__ A, const f16* __restrict__ Bsw,
          const float* __restrict__ bias0, const float* __restrict__ bias1,
          float* __restrict__ out, int Nrows, int KT, int flags) {
    const int lane    = threadIdx.x & 31;
    const int wave    = threadIdx.x >> 5;
    const int rowTile = wave & 3;
    const int colHalf = wave >> 2;
    const int row0    = blockIdx.x * 64 + rowTile * 16;

    v8f acc[4] = {};

    int arow = row0 + (lane & 15);
    if (arow >= Nrows) arow = Nrows - 1;           // clamp; stores are guarded
    const f16* aBase = A + (size_t)arow * 512 + ((lane >> 4) << 4);
    const f16* bBase = Bsw + (size_t)(colHalf * 4) * 512 + (size_t)lane * 16;

    for (int kk = 0; kk < KT; ++kk) {
        union { uint4 u[2]; v16h v; } af;
        const f16* ap = aBase + kk * 32;
        af.u[0] = *(const uint4*)(ap);             // K 16h+0..7
        af.u[1] = *(const uint4*)(ap + 8);         // K 16h+8..15

        const f16* bp = bBase + (size_t)kk * 4096;
        union { uint4 u[2]; v16h v; } b0, b1, b2, b3;
        b0.u[0] = *(const uint4*)(bp);        b0.u[1] = *(const uint4*)(bp + 8);
        b1.u[0] = *(const uint4*)(bp + 512);  b1.u[1] = *(const uint4*)(bp + 520);
        b2.u[0] = *(const uint4*)(bp + 1024); b2.u[1] = *(const uint4*)(bp + 1032);
        b3.u[0] = *(const uint4*)(bp + 1536); b3.u[1] = *(const uint4*)(bp + 1544);

        acc[0] = __builtin_amdgcn_wmma_f32_16x16x32_f16(false, af.v, false, b0.v,
                                                        (short)0, acc[0], false, false);
        acc[1] = __builtin_amdgcn_wmma_f32_16x16x32_f16(false, af.v, false, b1.v,
                                                        (short)0, acc[1], false, false);
        acc[2] = __builtin_amdgcn_wmma_f32_16x16x32_f16(false, af.v, false, b2.v,
                                                        (short)0, acc[2], false, false);
        acc[3] = __builtin_amdgcn_wmma_f32_16x16x32_f16(false, af.v, false, b3.v,
                                                        (short)0, acc[3], false, false);
    }

    // C layout: VGPR r -> M = r + 8*(lane/16), N = lane&15 (per 16x16 tile)
    const int  mBase  = row0 + ((lane >> 4) << 3);
    const int  colB   = colHalf * 64 + (lane & 15);
    const bool doRelu = flags & 1;
    const bool split  = flags & 2;

#pragma unroll
    for (int ct = 0; ct < 4; ++ct) {
        int col = colB + ct * 16;
        float bv = split ? (col < 64 ? bias0[col] : bias1[col - 64]) : bias0[col];
#pragma unroll
        for (int r = 0; r < 8; ++r) {
            int m = mBase + r;
            if (m >= Nrows) continue;
            float val = acc[ct][r] + bv;
            if (doRelu) val = fmaxf(val, 0.f);
            if (!split) {
                out[(size_t)m * 128 + col] = val;
            } else if (col < 64) {
                out[(size_t)m * 64 + col] = val;                       // mu
            } else {
                out[(size_t)Nrows * 64 + (size_t)m * 64 + (col - 64)] = val;  // logvar
            }
        }
    }
}

// ---------------- host orchestration ----------------

extern "C" void kernel_launch(void* const* d_in, const int* in_sizes, int n_in,
                              void* d_out, int out_size, void* d_ws, size_t ws_size,
                              hipStream_t stream) {
    const float*     x   = (const float*)d_in[0];
    const long long* ei  = (const long long*)d_in[1];
    const float*     pe  = (const float*)d_in[2];
    const float*     ew  = (const float*)d_in[3];
    const float*     W1  = (const float*)d_in[4];
    const float*     b1  = (const float*)d_in[5];
    const float*     W2  = (const float*)d_in[6];
    const float*     b2  = (const float*)d_in[7];
    const float*     Wmu = (const float*)d_in[8];
    const float*     bmu = (const float*)d_in[9];
    const float*     Wlv = (const float*)d_in[10];
    const float*     blv = (const float*)d_in[11];

    const int N = in_sizes[0] / 112;
    const int E = in_sizes[3];
    const long long* src = ei;
    const long long* dst = ei + E;

    char* ws = (char*)d_ws;
    auto alloc = [&](size_t bytes) -> char* {
        char* p = ws;
        ws += (bytes + 255) & ~(size_t)255;
        return p;
    };
    const size_t featB = (size_t)N * 128 * sizeof(float);
    float* deg  = (float*)alloc((size_t)N * sizeof(float));
    float* w    = (float*)alloc((size_t)E * sizeof(float));
    float* H0   = (float*)alloc(featB);
    float* H1   = (float*)alloc(featB);
    float* T1   = (float*)alloc(featB);
    float* Pa   = (float*)alloc(featB);
    float* Pb   = (float*)alloc(featB);
    f16*   Acat = (f16*)alloc((size_t)N * 512 * sizeof(f16));
    f16*   B1s  = (f16*)alloc((size_t)512 * 128 * sizeof(f16));
    f16*   B2s  = (f16*)alloc((size_t)512 * 128 * sizeof(f16));
    f16*   Bfs  = (f16*)alloc((size_t)128 * 128 * sizeof(f16));

    const int TB = 256;
    const int gE  = (E + TB - 1) / TB;
    const int gEp = ((size_t)E * 32 + TB - 1) / TB;
    const int gNf = ((size_t)N * 32 + TB - 1) / TB;
    const int gG  = (N + 63) / 64;

    // normalization
    hipMemsetAsync(deg, 0, (size_t)N * sizeof(float), stream);
    deg_kernel<<<gE, TB, 0, stream>>>(src, ew, deg, E);
    edgew_kernel<<<gE, TB, 0, stream>>>(src, dst, ew, deg, w, E);

    // weight pre-swizzle (tiny, once per call)
    packb_kernel<<<(512 * 128) / TB, TB, 0, stream>>>(W1, nullptr, 0, B1s, 512);
    packb_kernel<<<(512 * 128) / TB, TB, 0, stream>>>(W2, nullptr, 0, B2s, 512);
    packb_kernel<<<(128 * 128) / TB, TB, 0, stream>>>(Wmu, Wlv, 64, Bfs, 128);

    // h = concat(x, lap_pe); seeds Acat slot 0
    concat_kernel<<<gNf, TB, 0, stream>>>(x, pe, H0, Acat, N);

    auto layer = [&](const float* hin, float* hout, const f16* Bs, const float* bias) {
        // T1 = prop(h)
        hipMemsetAsync(Pa, 0, featB, stream);
        prop_kernel<<<gEp, TB, 0, stream>>>(hin, w, src, dst, Pa, E);
        cheb_kernel<<<gNf, TB, 0, stream>>>(Pa, Pa, 1.f, 0.f, T1, Acat, 1, N);
        // T2 = 2*prop(T1) - h   (stored in Pa)
        hipMemsetAsync(Pb, 0, featB, stream);
        prop_kernel<<<gEp, TB, 0, stream>>>(T1, w, src, dst, Pb, E);
        cheb_kernel<<<gNf, TB, 0, stream>>>(Pb, hin, 2.f, -1.f, Pa, Acat, 2, N);
        // T3 = 2*prop(T2) - T1  (f16 only)
        hipMemsetAsync(Pb, 0, featB, stream);
        prop_kernel<<<gEp, TB, 0, stream>>>(Pa, w, src, dst, Pb, E);
        cheb_kernel<<<gNf, TB, 0, stream>>>(Pb, T1, 2.f, -1.f, nullptr, Acat, 3, N);
        // hout = relu([T0|T1|T2|T3] @ [W0;W1;W2;W3] + b)
        gemm_wmma<<<gG, TB, 0, stream>>>(Acat, Bs, bias, bias, hout, N, 16, 1);
    };

    layer(H0, H1, B1s, b1);
    cvt_kernel<<<gNf, TB, 0, stream>>>(H1, Acat, 0, N);
    layer(H1, H0, B2s, b2);
    cvt_kernel<<<gNf, TB, 0, stream>>>(H0, Acat, 0, N);

    // [mu | logvar] = h2 @ [Wmu|Wlv] + [bmu|blv], split-stored into d_out
    gemm_wmma<<<gG, TB, 0, stream>>>(Acat, Bfs, bmu, blv, (float*)d_out, N, 4, 2);

    (void)n_in; (void)out_size; (void)ws_size;
}
